// DGCNN_59064390254918
// MI455X (gfx1250) — matrix operand order repaired and checked by reading, not compiled, not run on hardware
//
#include <hip/hip_runtime.h>
#include <hip/hip_bf16.h>

typedef __attribute__((ext_vector_type(16))) _Float16 v16h;
typedef __attribute__((ext_vector_type(8)))  _Float16 v8h;
typedef __attribute__((ext_vector_type(8)))  float    v8f;
typedef __attribute__((ext_vector_type(4)))  float    v4f;

#define KNN 20

__device__ __forceinline__ v8f wmma16(v16h a, v16h b, v8f c) {
  return __builtin_amdgcn_wmma_f32_16x16x32_f16(false, a, false, b, (short)0, c, false, false);
}

__device__ __forceinline__ int laneid() { return (int)(threadIdx.x & 31u); }

__device__ __forceinline__ v16h cat16(v8h lo, v8h hi) {
  return __builtin_shufflevector(lo, hi, 0, 1, 2, 3, 4, 5, 6, 7, 8, 9, 10, 11, 12, 13, 14, 15);
}

// A tile: 16 rows x 32 K from row-major f16 (global or LDS). All addresses 16B aligned
// by construction (ld multiple of 16 halfs, k0 multiple of 32, lane offset 16 halfs).
// Contiguous K order used here: lane<16 row=lane K k0+0..15; lane>=16 row=lane-16 K k0+16..31.
__device__ __forceinline__ v16h load_A(const _Float16* M, int ld, int row0, int k0) {
  int lane = laneid();
  const _Float16* p = M + (size_t)(row0 + (lane & 15)) * ld + k0 + ((lane >> 4) << 4);
  v8h lo = *(const v8h*)(p);
  v8h hi = *(const v8h*)(p + 8);
  return cat16(lo, hi);
}

// A tile with per-row gather through a flattened [N*KNN] index array.
__device__ __forceinline__ v16h load_A_gather(const _Float16* F, int ld, const int* gidx, int g0, int k0) {
  int lane = laneid();
  int src = gidx[g0 + (lane & 15)];
  const _Float16* p = F + (size_t)src * ld + k0 + ((lane >> 4) << 4);
  v8h lo = *(const v8h*)(p);
  v8h hi = *(const v8h*)(p + 8);
  return cat16(lo, hi);
}

// B tile for Gram: B = F^T, B[k][c] = F[c][k]; 32 contiguous bytes per lane.
// lane<16 col=lane K k0+0..15; lane>=16 col=lane-16 K k0+16..31.
__device__ __forceinline__ v16h load_B_T(const _Float16* F, int ld, int col0, int k0) {
  int lane = laneid();
  const _Float16* p = F + (size_t)(col0 + (lane & 15)) * ld + k0 + ((lane >> 4) << 4);
  v8h lo = *(const v8h*)(p);
  v8h hi = *(const v8h*)(p + 8);
  return cat16(lo, hi);
}

// Pre-swizzled weight B tile: per (ktile,ctile) each lane's 16 halfs are contiguous.
__device__ __forceinline__ v16h load_B_swz(const _Float16* Wsz, int CT, int kt, int ct) {
  int lane = laneid();
  const _Float16* p = Wsz + ((size_t)((kt * CT + ct) * 32 + lane) << 4);
  v8h lo = *(const v8h*)(p);
  v8h hi = *(const v8h*)(p + 8);
  return cat16(lo, hi);
}

// ---------------- prep: X f32 [N,9] -> Xh f16 [N,32] zero-padded, sq = |x|^2 ----------------
__global__ void k_prep_x(const float* __restrict__ X, _Float16* __restrict__ Xh,
                         float* __restrict__ sq, int n) {
  int i = blockIdx.x * blockDim.x + threadIdx.x;
  if (i >= n) return;
  float s = 0.f;
#pragma unroll
  for (int d = 0; d < 32; ++d) {
    float v = (d < 9) ? X[(size_t)i * 9 + d] : 0.f;
    s += v * v;
    Xh[(size_t)i * 32 + d] = (_Float16)v;
  }
  sq[i] = s;
}

// ------------- weights: f32 row-major [r,c] -> f16 zero-padded [R,C] in WMMA-swizzled tile order -------------
__global__ void k_cvt_swz(const float* __restrict__ W, _Float16* __restrict__ Wsz,
                          int r, int c, int C, int total) {
  int o = blockIdx.x * blockDim.x + threadIdx.x;
  if (o >= total) return;
  int i = o & 15;
  int l = (o >> 4) & 31;
  int t = o >> 9;
  int CT = C >> 4;
  int kt = t / CT, ct = t % CT;
  int col = ct * 16 + (l & 15);
  int k   = kt * 32 + ((l >> 4) << 4) + i;
  float v = (k < r && col < c) ? W[(size_t)k * c + col] : 0.f;
  Wsz[o] = (_Float16)v;
}

// ---------------- fused kNN: Gram via WMMA, top-20 merge pipelined into the WMMA shadow ----------------
// Column block = 32 (two 16x16 tiles). Iteration t: issue WMMAs(t), merge candidates of t-1
// (VALU+DS, co-executes with in-flight 16-bit WMMA per ISA), then store distance tile t.
template <int KP>
__global__ void __launch_bounds__(32) k_knn(const _Float16* __restrict__ F,
                                            const float* __restrict__ sq,
                                            int* __restrict__ idx, int n) {
  __shared__ float topd[32][KNN];      // lane's list: row = lane&15, col-half = lane>>4
  __shared__ int   topi[32][KNN];
  __shared__ float dtile[32][16];      // [cand col in block][query row]
  int lane = laneid();
  int row0 = blockIdx.x * 16;

#pragma unroll
  for (int j = 0; j < KNN; ++j) { topd[lane][j] = 3.0e38f; topi[lane][j] = 0; }
  float thr = 3.0e38f;

  v16h a0 = load_A(F, KP, row0, 0);
  v16h a1;
  if constexpr (KP == 64) a1 = load_A(F, KP, row0, 32);
  __syncthreads();

  const int row = lane & 15;
  const int j0  = (lane >> 4) << 3;
  const int c   = lane & 15;
  const int rb  = (lane >> 4) << 3;

  // merge 16 candidates (2 x 8) of block starting at colbase from dtile into this lane's list
  auto merge = [&](int colbase) {
#pragma unroll
    for (int h = 0; h < 2; ++h) {
#pragma unroll 1
      for (int j = 0; j < 8; ++j) {
        float d = dtile[h * 16 + j0 + j][row];
        if (d < thr) {
          int cand = colbase + h * 16 + j0 + j;
          int p = KNN - 1;
          while (p > 0 && topd[lane][p - 1] > d) {
            topd[lane][p] = topd[lane][p - 1];
            topi[lane][p] = topi[lane][p - 1];
            --p;
          }
          topd[lane][p] = d; topi[lane][p] = cand;
          thr = topd[lane][KNN - 1];
        }
      }
    }
  };

  for (int col0 = 0; col0 < n; col0 += 32) {
    if (col0 + 32 < n) __builtin_prefetch(F + (size_t)(col0 + 32) * KP, 0, 0);
    // issue WMMAs for current block (accumulators stay in flight across the merge)
    v8f accA = {}, accB = {};
    accA = wmma16(a0, load_B_T(F, KP, col0, 0), accA);
    accB = wmma16(a0, load_B_T(F, KP, col0 + 16, 0), accB);
    if constexpr (KP == 64) {
      accA = wmma16(a1, load_B_T(F, KP, col0, 32), accA);
      accB = wmma16(a1, load_B_T(F, KP, col0 + 16, 32), accB);
    }
    // merge previous block while the WMMAs execute
    if (col0 > 0) merge(col0 - 32);
    __syncthreads();
    // surrogate distance d = sq[c] - 2<q,c> (row term constant -> same ranking)
    float scA = sq[col0 + c];
    float scB = sq[col0 + 16 + c];
    v4f lA, hA, lB, hB;
#pragma unroll
    for (int i = 0; i < 4; ++i) {
      lA[i] = scA - 2.0f * accA[i];
      hA[i] = scA - 2.0f * accA[4 + i];
      lB[i] = scB - 2.0f * accB[i];
      hB[i] = scB - 2.0f * accB[4 + i];
    }
    *(v4f*)&dtile[c][rb]          = lA;
    *(v4f*)&dtile[c][rb + 4]      = hA;
    *(v4f*)&dtile[16 + c][rb]     = lB;
    *(v4f*)&dtile[16 + c][rb + 4] = hB;
    __syncthreads();
  }
  merge(n - 32);
  __syncthreads();

  // merge the two sorted half-lists per row -> global top-20
  if (lane < 16) {
    int ia = 0, ib = 0;
#pragma unroll
    for (int j = 0; j < KNN; ++j) {
      float da = topd[lane][ia];
      float db = topd[lane + 16][ib];
      bool  sb = (db < da);
      idx[(size_t)(row0 + lane) * KNN + j] = sb ? topi[lane + 16][ib] : topi[lane][ia];
      ia += sb ? 0 : 1;
      ib += sb ? 1 : 0;
    }
  }
}

// ------------- block: gathered [16 pts * 20 nb] @ W[KP,64], max over nb, +bias, fused |.|^2 -------------
template <int KP>
__global__ void __launch_bounds__(32) k_block(const _Float16* __restrict__ F,
                                              const int* __restrict__ idx,
                                              const _Float16* __restrict__ Wsz,
                                              const float* __restrict__ bias,
                                              _Float16* __restrict__ Oh,
                                              float* __restrict__ sqo, int n) {
  __shared__ _Float16 res[64][16 * KNN];   // [out col][gathered row], 40 KB
  __shared__ float part[16][32];
  int lane = laneid();
  int p0 = blockIdx.x * 16;
  int g0 = p0 * KNN;
#pragma unroll
  for (int p = 0; p < 16; ++p) part[p][lane] = 0.f;

  for (int at = 0; at < KNN; ++at) {
    v16h a0 = load_A_gather(F, KP, idx, g0 + at * 16, 0);
    v16h a1;
    if constexpr (KP == 64) a1 = load_A_gather(F, KP, idx, g0 + at * 16, 32);
#pragma unroll
    for (int ct = 0; ct < 4; ++ct) {
      v8f acc = {};
      acc = wmma16(a0, load_B_swz(Wsz, 4, 0, ct), acc);
      if constexpr (KP == 64) acc = wmma16(a1, load_B_swz(Wsz, 4, 1, ct), acc);
      int col = ct * 16 + (lane & 15);
      int rb  = at * 16 + ((lane >> 4) << 3);
      v8h h;
#pragma unroll
      for (int i = 0; i < 8; ++i) h[i] = (_Float16)acc[i];
      *(v8h*)&res[col][rb] = h;    // contiguous 16B per lane
    }
  }
  __syncthreads();
  // max over 20 neighbors, add bias; deterministic per-lane partial sums for |row|^2
  for (int i = lane; i < 16 * 64; i += 32) {
    int p = i >> 6, cc = i & 63;
    float m = -3.0e38f;
#pragma unroll 1
    for (int t = 0; t < KNN; ++t) m = fmaxf(m, (float)res[cc][p * KNN + t]);
    m += bias[cc];
    Oh[(size_t)(p0 + p) * 64 + cc] = (_Float16)m;
    part[p][lane] += m * m;
  }
  __syncthreads();
  if (lane < 16 && sqo != nullptr) {
    float s = 0.f;
#pragma unroll
    for (int j = 0; j < 32; ++j) s += part[lane][j];
    sqo[p0 + lane] = s;
  }
}

// bias (+optional relu) + f16 store of one C tile into LDS buffer
__device__ __forceinline__ void store_tile(_Float16* Bf, int ld, int ct, const v8f& acc,
                                           const float* bias, bool relu) {
  int lane = laneid();
  int col = ct * 16 + (lane & 15);
  int rb  = (lane >> 4) << 3;
  float bb = bias[col];
#pragma unroll
  for (int i = 0; i < 8; ++i) {
    float v = acc[i] + bb;
    if (relu) v = fmaxf(v, 0.f);
    Bf[(size_t)(rb + i) * ld + col] = (_Float16)v;
  }
}

// ------------- fused head: maxpool -> fc96_1024 -> concat Z[1216] -> 3x relu-fc -> fc128_3 -------------
__global__ void __launch_bounds__(32) k_head(
    const _Float16* __restrict__ X1, const _Float16* __restrict__ X2, const _Float16* __restrict__ X3,
    const _Float16* __restrict__ w3s,  const float* __restrict__ b3,
    const _Float16* __restrict__ wc1s, const float* __restrict__ bc1,
    const _Float16* __restrict__ wc2s, const float* __restrict__ bc2,
    const _Float16* __restrict__ wc3s, const float* __restrict__ bc3,
    const _Float16* __restrict__ wc4s, const float* __restrict__ bc4,
    float* __restrict__ out, int n) {
  __shared__ _Float16 Z[16][1216];   // 38 KB: [X1|X2|X3|H]
  __shared__ _Float16 P[16][96];
  __shared__ _Float16 B1[16][256];
  __shared__ _Float16 B2[16][256];
  __shared__ _Float16 B3[16][128];
  int lane = laneid();
  int p0 = blockIdx.x * 16;

  for (int i = lane; i < 16 * 64; i += 32) {
    int p = i >> 6, c = i & 63;
    Z[p][c]       = X1[(size_t)(p0 + p) * 64 + c];
    Z[p][64 + c]  = X2[(size_t)(p0 + p) * 64 + c];
    Z[p][128 + c] = X3[(size_t)(p0 + p) * 64 + c];
  }
  __syncthreads();
  for (int i = lane; i < 16 * 96; i += 32) {
    int p = i / 96, j = i % 96;
    P[p][j] = (_Float16)fmaxf((float)Z[p][2 * j], (float)Z[p][2 * j + 1]);
  }
  __syncthreads();
  // H = P @ w3 + b3 -> Z[:,192:1216]; A tiles live in registers
  {
    v16h a0 = load_A(&P[0][0], 96, 0, 0);
    v16h a1 = load_A(&P[0][0], 96, 0, 32);
    v16h a2 = load_A(&P[0][0], 96, 0, 64);
    for (int ctg = 0; ctg < 16; ++ctg) {
#pragma unroll
      for (int u = 0; u < 4; ++u) {
        int ct = ctg * 4 + u;
        v8f acc = {};
        acc = wmma16(a0, load_B_swz(w3s, 64, 0, ct), acc);
        acc = wmma16(a1, load_B_swz(w3s, 64, 1, ct), acc);
        acc = wmma16(a2, load_B_swz(w3s, 64, 2, ct), acc);
        store_tile(&Z[0][192], 1216, ct, acc, b3, false);
      }
    }
  }
  __syncthreads();
  // L1: B1 = relu(Z @ wc1 + bc1), K=1216 (38 ktiles), 256 cols, ct blocked x4
  for (int ctg = 0; ctg < 4; ++ctg) {
    v8f c0 = {}, c1 = {}, c2 = {}, c3 = {};
    for (int kt = 0; kt < 38; ++kt) {
      v16h a = load_A(&Z[0][0], 1216, 0, kt * 32);
      c0 = wmma16(a, load_B_swz(wc1s, 16, kt, ctg * 4 + 0), c0);
      c1 = wmma16(a, load_B_swz(wc1s, 16, kt, ctg * 4 + 1), c1);
      c2 = wmma16(a, load_B_swz(wc1s, 16, kt, ctg * 4 + 2), c2);
      c3 = wmma16(a, load_B_swz(wc1s, 16, kt, ctg * 4 + 3), c3);
    }
    store_tile(&B1[0][0], 256, ctg * 4 + 0, c0, bc1, true);
    store_tile(&B1[0][0], 256, ctg * 4 + 1, c1, bc1, true);
    store_tile(&B1[0][0], 256, ctg * 4 + 2, c2, bc1, true);
    store_tile(&B1[0][0], 256, ctg * 4 + 3, c3, bc1, true);
  }
  __syncthreads();
  // L2: B2 = relu(B1 @ wc2 + bc2), K=256
  for (int ctg = 0; ctg < 4; ++ctg) {
    v8f c0 = {}, c1 = {}, c2 = {}, c3 = {};
    for (int kt = 0; kt < 8; ++kt) {
      v16h a = load_A(&B1[0][0], 256, 0, kt * 32);
      c0 = wmma16(a, load_B_swz(wc2s, 16, kt, ctg * 4 + 0), c0);
      c1 = wmma16(a, load_B_swz(wc2s, 16, kt, ctg * 4 + 1), c1);
      c2 = wmma16(a, load_B_swz(wc2s, 16, kt, ctg * 4 + 2), c2);
      c3 = wmma16(a, load_B_swz(wc2s, 16, kt, ctg * 4 + 3), c3);
    }
    store_tile(&B2[0][0], 256, ctg * 4 + 0, c0, bc2, true);
    store_tile(&B2[0][0], 256, ctg * 4 + 1, c1, bc2, true);
    store_tile(&B2[0][0], 256, ctg * 4 + 2, c2, bc2, true);
    store_tile(&B2[0][0], 256, ctg * 4 + 3, c3, bc2, true);
  }
  __syncthreads();
  // L3: B3 = relu(B2 @ wc3 + bc3), K=256, 128 cols
  for (int ctg = 0; ctg < 2; ++ctg) {
    v8f c0 = {}, c1 = {}, c2 = {}, c3 = {};
    for (int kt = 0; kt < 8; ++kt) {
      v16h a = load_A(&B2[0][0], 256, 0, kt * 32);
      c0 = wmma16(a, load_B_swz(wc3s, 8, kt, ctg * 4 + 0), c0);
      c1 = wmma16(a, load_B_swz(wc3s, 8, kt, ctg * 4 + 1), c1);
      c2 = wmma16(a, load_B_swz(wc3s, 8, kt, ctg * 4 + 2), c2);
      c3 = wmma16(a, load_B_swz(wc3s, 8, kt, ctg * 4 + 3), c3);
    }
    store_tile(&B3[0][0], 128, ctg * 4 + 0, c0, bc3, true);
    store_tile(&B3[0][0], 128, ctg * 4 + 1, c1, bc3, true);
    store_tile(&B3[0][0], 128, ctg * 4 + 2, c2, bc3, true);
    store_tile(&B3[0][0], 128, ctg * 4 + 3, c3, bc3, true);
  }
  __syncthreads();
  // L4: out = B3 @ wc4 + bc4, K=128, cols padded 3->16
  {
    v8f acc = {};
#pragma unroll
    for (int kt = 0; kt < 4; ++kt)
      acc = wmma16(load_A(&B3[0][0], 128, 0, kt * 32), load_B_swz(wc4s, 1, kt, 0), acc);
    int col = lane & 15;
    int rb  = (lane >> 4) << 3;
    if (col < 3) {
      float bb = bc4[col];
#pragma unroll
      for (int i = 0; i < 8; ++i) out[(size_t)(p0 + rb + i) * 3 + col] = acc[i] + bb;
    }
  }
}

extern "C" void kernel_launch(void* const* d_in, const int* in_sizes, int n_in,
                              void* d_out, int out_size, void* d_ws, size_t ws_size,
                              hipStream_t stream) {
  const float* X   = (const float*)d_in[0];
  const float* w1  = (const float*)d_in[1];
  const float* b1  = (const float*)d_in[2];
  const float* w2  = (const float*)d_in[3];
  const float* b2  = (const float*)d_in[4];
  const float* w3  = (const float*)d_in[5];
  const float* b3  = (const float*)d_in[6];
  const float* wc1 = (const float*)d_in[7];
  const float* bc1 = (const float*)d_in[8];
  const float* wc2 = (const float*)d_in[9];
  const float* bc2 = (const float*)d_in[10];
  const float* wc3 = (const float*)d_in[11];
  const float* bc3 = (const float*)d_in[12];
  const float* wc4 = (const float*)d_in[13];
  const float* bc4 = (const float*)d_in[14];
  float* out = (float*)d_out;

  const int N = in_sizes[0] / 9;

  char* ws = (char*)d_ws;
  auto alloc = [&](size_t bytes) -> void* {
    void* p = (void*)ws;
    ws += (bytes + 255) & ~(size_t)255;
    return p;
  };
  _Float16* Xh   = (_Float16*)alloc((size_t)N * 32 * 2);
  _Float16* X1h  = (_Float16*)alloc((size_t)N * 64 * 2);
  _Float16* X2h  = (_Float16*)alloc((size_t)N * 64 * 2);
  _Float16* X3h  = (_Float16*)alloc((size_t)N * 64 * 2);
  float*    sq   = (float*)alloc((size_t)N * 4);
  int*      idx  = (int*)alloc((size_t)N * KNN * 4);
  _Float16* w1s  = (_Float16*)alloc((size_t)32 * 64 * 2);
  _Float16* w2s  = (_Float16*)alloc((size_t)64 * 64 * 2);
  _Float16* w3s  = (_Float16*)alloc((size_t)96 * 1024 * 2);
  _Float16* wc1s = (_Float16*)alloc((size_t)1216 * 256 * 2);
  _Float16* wc2s = (_Float16*)alloc((size_t)256 * 256 * 2);
  _Float16* wc3s = (_Float16*)alloc((size_t)256 * 128 * 2);
  _Float16* wc4s = (_Float16*)alloc((size_t)128 * 16 * 2);

  auto cvt = [&](const float* src, _Float16* dst, int r, int c, int R, int C) {
    int total = R * C;
    k_cvt_swz<<<(total + 255) / 256, 256, 0, stream>>>(src, dst, r, c, C, total);
  };

  k_prep_x<<<(N + 255) / 256, 256, 0, stream>>>(X, Xh, sq, N);
  cvt(w1, w1s, 9, 64, 32, 64);
  cvt(w2, w2s, 64, 64, 64, 64);
  cvt(w3, w3s, 96, 1024, 96, 1024);
  cvt(wc1, wc1s, 1216, 256, 1216, 256);
  cvt(wc2, wc2s, 256, 256, 256, 256);
  cvt(wc3, wc3s, 256, 128, 256, 128);
  cvt(wc4, wc4s, 128, 3, 128, 16);

  dim3 tiles(N / 16), wave(32);

  // Block 1
  k_knn<32><<<tiles, wave, 0, stream>>>(Xh, sq, idx, N);
  k_block<32><<<tiles, wave, 0, stream>>>(Xh, idx, w1s, b1, X1h, sq, N);
  // Block 2
  k_knn<64><<<tiles, wave, 0, stream>>>(X1h, sq, idx, N);
  k_block<64><<<tiles, wave, 0, stream>>>(X1h, idx, w2s, b2, X2h, sq, N);
  // Block 3
  k_knn<64><<<tiles, wave, 0, stream>>>(X2h, sq, idx, N);
  k_block<64><<<tiles, wave, 0, stream>>>(X2h, idx, w2s, b2, X3h, nullptr, N);
  // Head
  k_head<<<tiles, wave, 0, stream>>>(X1h, X2h, X3h, w3s, b3, wc1s, bc1,
                                     wc2s, bc2, wc3s, bc3, wc4s, bc4, out, N);
}
